// GRU_38044820308380
// MI455X (gfx1250) — matrix operand order repaired and checked by reading, not compiled, not run on hardware
//
#include <hip/hip_runtime.h>

// ---------------------------------------------------------------------------
// Persistent-GRU for MI455X (gfx1250, wave32, WMMA f32<-f16 16x16x32).
//
//   T=2048, B=64, D=256, H=512, DH=768. ~309 GFLOP, recurrence-latency bound.
//   Weights pre-swizzled to f16 WMMA B-fragment order, register-resident per
//   wave (24 x v16h) for all 2048 steps. h carried in fp32 (hbuf); f16 shadow
//   copies (hbuf16, rhbuf16) feed the matmul so A-assembly is a pure copy.
//
//   Grid: 32 persistent WGs x 384 threads (12 waves). WG j owns H columns
//   [16j,16j+16). waves 0-3: z, 4-7: r (+ r*h f16 publish), 8-11: h~ + update.
//   Two L2-atomic grid barriers per step. gemm768 uses 4 independent WMMA
//   accumulator chains for ILP (avoids C-operand RAW serialization).
// ---------------------------------------------------------------------------

typedef __attribute__((ext_vector_type(16))) _Float16 v16h;
typedef __attribute__((ext_vector_type(8)))  _Float16 v8h;
typedef __attribute__((ext_vector_type(4)))  _Float16 v4h;
typedef __attribute__((ext_vector_type(8)))  float    v8f;
typedef __attribute__((ext_vector_type(4)))  float    v4f;

#define T_STEPS 2048
#define BB      64
#define DD      256
#define HH      512
#define DHK     768
#define NWG     32
#define NTHREADS 384          // 12 waves
#define KT      24            // 768 / 32 K-tiles
#define ASTRIDE 776           // 768 + 8 halves pad -> conflict-free b128 reads

// ---------------- device helpers ----------------

__device__ __forceinline__ void grid_sync(int* cnt, int target) {
  __threadfence();
  __syncthreads();
  if (threadIdx.x == 0) {
    __hip_atomic_fetch_add(cnt, 1, __ATOMIC_RELEASE, __HIP_MEMORY_SCOPE_AGENT);
    while (__hip_atomic_load(cnt, __ATOMIC_ACQUIRE, __HIP_MEMORY_SCOPE_AGENT) < target)
      __builtin_amdgcn_s_sleep(1);
  }
  __syncthreads();
}

__device__ __forceinline__ v16h loadA(const _Float16* A, int base) {
  v8h lo = *(const v8h*)(A + base);        // K regs 0..3
  v8h hi = *(const v8h*)(A + base + 16);   // K regs 4..7
  return __builtin_shufflevector(lo, hi, 0,1,2,3,4,5,6,7,
                                         8,9,10,11,12,13,14,15);
}

// K=768 GEMM: 24 wmma split into 4 independent accumulation chains (ILP).
__device__ __forceinline__ v8f gemm768(const _Float16* A, int arow, int lhi,
                                       const v16h* wB, float bias) {
  v8f c0, c1, c2, c3;
  #pragma unroll
  for (int v = 0; v < 8; ++v) { c0[v] = bias; c1[v] = 0.f; c2[v] = 0.f; c3[v] = 0.f; }
  #pragma unroll
  for (int k = 0; k < KT; k += 4) {
    const int b0 = arow + (k + 0) * 32 + 8 * lhi;
    const int b1 = arow + (k + 1) * 32 + 8 * lhi;
    const int b2 = arow + (k + 2) * 32 + 8 * lhi;
    const int b3 = arow + (k + 3) * 32 + 8 * lhi;
    v16h a0 = loadA(A, b0), a1 = loadA(A, b1), a2 = loadA(A, b2), a3 = loadA(A, b3);
    c0 = __builtin_amdgcn_wmma_f32_16x16x32_f16(false, a0, false, wB[k+0], (short)0, c0, false, false);
    c1 = __builtin_amdgcn_wmma_f32_16x16x32_f16(false, a1, false, wB[k+1], (short)0, c1, false, false);
    c2 = __builtin_amdgcn_wmma_f32_16x16x32_f16(false, a2, false, wB[k+2], (short)0, c2, false, false);
    c3 = __builtin_amdgcn_wmma_f32_16x16x32_f16(false, a3, false, wB[k+3], (short)0, c3, false, false);
  }
  c0 = c0 + c1;
  c2 = c2 + c3;
  return c0 + c2;
}

__device__ __forceinline__ float fast_sigmoid(float x) {
  return 1.f / (1.f + __expf(-x));
}
__device__ __forceinline__ float fast_tanh(float x) {
  // 1 - 2/(e^{2x}+1): saturates cleanly at +-1, no inf/inf.
  return 1.f - 2.f / (__expf(2.f * x) + 1.f);
}

// ---------------- setup kernels ----------------

__global__ void gru_init_kernel(const float* __restrict__ h0,
                                float* __restrict__ hbuf,
                                _Float16* __restrict__ hbuf16, int* cnt) {
  if (blockIdx.x == 0 && threadIdx.x == 0) *cnt = 0;
  for (int i = threadIdx.x + blockIdx.x * blockDim.x; i < BB * HH;
       i += blockDim.x * gridDim.x) {
    float h = h0[i];
    hbuf[i] = h;
    hbuf16[i] = (_Float16)h;
  }
}

// fp32 [768,512] -> f16, swizzled to WMMA B-fragment order:
//   fragment (g, cblk, ktile): 32 lanes x 16 halves.
//   lane l: N = cblk*16 + (l&15); K-local = (i<8 ? 8*(l>>4)+i : 16+8*(l>>4)+(i-8)), i = half idx.
__global__ void gru_convert_kernel(const float* __restrict__ Wz,
                                   const float* __restrict__ Wr,
                                   const float* __restrict__ Wh,
                                   _Float16* __restrict__ wsw) {
  const int total = 3 * 32 * KT * 32 * 16;
  int n = blockIdx.x * blockDim.x + threadIdx.x;
  if (n >= total) return;
  int i    = n & 15;
  int l    = (n >> 4) & 31;
  int frag = n >> 9;            // ((g*32 + c)*KT + kt)
  int kt   = frag % KT;
  int cg   = frag / KT;
  int c    = cg & 31;
  int g    = cg >> 5;
  int col    = c * 16 + (l & 15);
  int Klocal = ((i >> 3) << 4) + ((l >> 4) << 3) + (i & 7);
  int K      = kt * 32 + Klocal;
  const float* W = (g == 0) ? Wz : ((g == 1) ? Wr : Wh);
  wsw[n] = (_Float16)W[K * HH + col];
}

// ---------------- persistent GRU kernel ----------------

__global__ void __launch_bounds__(NTHREADS, 1)
gru_persistent_kernel(const float* __restrict__ x,
                      const float* __restrict__ bz,
                      const float* __restrict__ br,
                      const float* __restrict__ bh,
                      const _Float16* __restrict__ wsw,
                      float* __restrict__ hbuf,
                      _Float16* __restrict__ hbuf16,
                      _Float16* __restrict__ rhbuf16,
                      int* cnt,
                      float* __restrict__ out) {
  __shared__ _Float16 Alds[BB * ASTRIDE];       // [64 x 768(+pad)] f16 xh
  __shared__ float    zbuf[4 * 16 * 16];        // z tiles handoff (waves 0-3 -> 8-11)

  const int tid  = threadIdx.x;
  const int wave = tid >> 5;
  const int lane = tid & 31;
  const int cblk = blockIdx.x;                  // 0..31, owns H cols [16c,16c+16)
  const int l15  = lane & 15;
  const int lhi  = lane >> 4;
  const int mt   = wave & 3;                    // M tile (rows 16*mt..)
  const int gate = wave >> 2;                   // 0=z, 1=r, 2=h~

  // Load this wave's [768x16] f16 weight slice into registers (24 x v16h).
  v16h wB[KT];
  {
    const v16h* wp = (const v16h*)wsw + ((size_t)(gate * 32 + cblk) * KT) * 32 + lane;
    #pragma unroll
    for (int k = 0; k < KT; ++k) wB[k] = wp[k * 32];
  }
  const float bias = (gate == 0 ? bz : (gate == 1 ? br : bh))[cblk * 16 + l15];
  const int arow = (16 * mt + l15) * ASTRIDE;

  int target = NWG;
  #pragma unroll 1
  for (int t = 0; t < T_STEPS; ++t) {
    // ---- assemble A = [x_t | h_{t-1}] in LDS (f16) ----
    {
      const float* xt = x + (size_t)t * (BB * DD);
      for (int q = tid; q < (BB * DD) / 4; q += NTHREADS) {   // x: f32 -> f16
        int row = q >> 6, c4 = (q & 63) << 2;
        v4f v = ((const v4f*)xt)[q];
        v4h h4 = {(_Float16)v.x, (_Float16)v.y, (_Float16)v.z, (_Float16)v.w};
        *(v4h*)(&Alds[row * ASTRIDE + c4]) = h4;
      }
      for (int q = tid; q < (BB * HH) / 8; q += NTHREADS) {   // h: f16 copy
        int row = q >> 6, c8 = (q & 63) << 3;
        *(v8h*)(&Alds[row * ASTRIDE + DD + c8]) = ((const v8h*)hbuf16)[q];
      }
      if (t + 1 < T_STEPS) {                    // global_prefetch next x slice
        const float* xn = x + (size_t)(t + 1) * (BB * DD);
        for (int q = tid * 64; q < BB * DD; q += NTHREADS * 64)
          __builtin_prefetch(xn + q, 0, 1);
      }
    }
    __syncthreads();

    // ---- phase A: z (waves 0-3) and r -> r*h (waves 4-7) ----
    if (gate == 0) {
      v8f c = gemm768(Alds, arow, lhi, wB, bias);
      #pragma unroll
      for (int v = 0; v < 8; ++v)
        zbuf[mt * 256 + (v + 8 * lhi) * 16 + l15] = fast_sigmoid(c[v]);
    } else if (gate == 1) {
      v8f c = gemm768(Alds, arow, lhi, wB, bias);
      #pragma unroll
      for (int v = 0; v < 8; ++v) {
        float r  = fast_sigmoid(c[v]);
        int row  = 16 * mt + v + 8 * lhi;
        int col  = cblk * 16 + l15;
        rhbuf16[row * HH + col] = (_Float16)(r * hbuf[row * HH + col]);
      }
    }
    grid_sync(cnt, target); target += NWG;      // r*h published device-wide

    // ---- reassemble h-part of A with r*h (f16 copy) ----
    for (int q = tid; q < (BB * HH) / 8; q += NTHREADS) {
      int row = q >> 6, c8 = (q & 63) << 3;
      *(v8h*)(&Alds[row * ASTRIDE + DD + c8]) = ((const v8h*)rhbuf16)[q];
    }
    __syncthreads();

    // ---- phase B: h~ and state update (waves 8-11) ----
    if (gate == 2) {
      v8f c = gemm768(Alds, arow, lhi, wB, bias);
      #pragma unroll
      for (int v = 0; v < 8; ++v) {
        float ht = fast_tanh(c[v]);
        int row  = 16 * mt + v + 8 * lhi;
        int col  = cblk * 16 + l15;
        float z  = zbuf[mt * 256 + (v + 8 * lhi) * 16 + l15];
        float hp = hbuf[row * HH + col];
        float hn = hp + z * (ht - hp);          // (1-z)h + z h~
        out[((size_t)t * BB + row) * HH + col] = hn;
        hbuf[row * HH + col] = hn;
        hbuf16[row * HH + col] = (_Float16)hn;
      }
    }
    grid_sync(cnt, target); target += NWG;      // h_t published device-wide
  }
}

// ---------------- launch ----------------

extern "C" void kernel_launch(void* const* d_in, const int* in_sizes, int n_in,
                              void* d_out, int out_size, void* d_ws, size_t ws_size,
                              hipStream_t stream) {
  const float* x  = (const float*)d_in[0];
  const float* h0 = (const float*)d_in[1];
  const float* Wz = (const float*)d_in[2];
  const float* bz = (const float*)d_in[3];
  const float* Wr = (const float*)d_in[4];
  const float* br = (const float*)d_in[5];
  const float* Wh = (const float*)d_in[6];
  const float* bh = (const float*)d_in[7];
  float* out = (float*)d_out;

  char* ws = (char*)d_ws;
  int*      cnt     = (int*)ws;                                   // 256 B
  float*    hbuf    = (float*)(ws + 256);                         // 128 KB f32
  _Float16* hbuf16  = (_Float16*)(ws + 256 + BB * HH * 4);        // 64 KB f16
  _Float16* rhbuf16 = (_Float16*)(ws + 256 + BB * HH * 6);        // 64 KB f16
  _Float16* wsw     = (_Float16*)(ws + 256 + BB * HH * 8);        // 2.36 MB f16

  gru_init_kernel<<<64, 256, 0, stream>>>(h0, hbuf, hbuf16, cnt);
  const int total = 3 * 32 * KT * 32 * 16;
  gru_convert_kernel<<<(total + 255) / 256, 256, 0, stream>>>(Wz, Wr, Wh, wsw);
  gru_persistent_kernel<<<NWG, NTHREADS, 0, stream>>>(x, bz, br, bh, wsw,
                                                      hbuf, hbuf16, rhbuf16,
                                                      cnt, out);
}
